// SpatialCrossAttention_60919816126555
// MI455X (gfx1250) — compile-verified, matrix operand order checked by reading, not run on hardware
//
#include <hip/hip_runtime.h>
#include <hip/hip_bf16.h>

// ---------------------------------------------------------------------------
// BEVFormer SpatialCrossAttention for MI455X (gfx1250, wave32).
//   B=1, S=6, N=10000, Z=4, C=128, H=48, W=88, M=4224, NH=4, NP=8, HD=32
// Pipeline:
//   0) pack weights -> f16 WMMA fragment order (one-time, tiny)
//   1) valproj = value @ Wv + bv                       (WMMA f16->f32)
//   2) projq   = (query+query_pos) @ [Wo|Wa] + [bo|ba] (fused WMMA, NCOLS=96)
//   3) sampler: per-query wave32 bilinear gather, scalar-broadcast weights
//   4) out     = slots @ Wp + bp + query (residual)    (WMMA)
// ---------------------------------------------------------------------------

#define S_ 6
#define N_ 10000
#define C_ 128
#define H_ 48
#define W_ 88
#define M_ (H_ * W_)   // 4224
#define NH_ 4
#define NP_ 8
#define HD_ 32

typedef __attribute__((ext_vector_type(16))) _Float16 v16h;
typedef __attribute__((ext_vector_type(8)))  _Float16 v8h;
typedef __attribute__((ext_vector_type(4)))  _Float16 v4h;
typedef __attribute__((ext_vector_type(8)))  float    v8f;

// CDNA5 ISA 7.12.2 16-bit fragment K map: element e of lane L covers
//   K = ((e&8)<<1) + (e&7) + (L>=16 ? 8 : 0)   within a 32-wide K chunk.

// ---------------------------------------------------------------------------
// Pack a weight matrix W (128 x ncols, f32 row-major) into f16 fragments:
//   dst[((ntile*4 + kc)*32 + lane)*16 + e]  (each lane's 16 halves contiguous,
//   each 16-col tile occupies a contiguous 2048-half block -> concatenable).
// ---------------------------------------------------------------------------
__global__ void pack_weights_kernel(const float* __restrict__ W,
                                    _Float16* __restrict__ dst, int ncols)
{
    const int gid = blockIdx.x * blockDim.x + threadIdx.x;   // < 128*ncols
    const int e     = gid & 15;
    const int lane  = (gid >> 4) & 31;
    const int kc    = (gid >> 9) & 3;
    const int ntile = gid >> 11;
    const int k = kc * 32 + ((e & 8) << 1) + (e & 7) + ((lane >= 16) ? 8 : 0);
    const int n = ntile * 16 + (lane & 15);
    dst[gid] = (_Float16)W[(size_t)k * ncols + n];
}

// Concatenate bo(64) | ba(32) into one 96-entry bias vector.
__global__ void concat_bias_kernel(const float* __restrict__ b0,
                                   const float* __restrict__ b1,
                                   float* __restrict__ dst)
{
    const int i = threadIdx.x;              // blockDim = 96
    dst[i] = (i < 64) ? b0[i] : b1[i - 64];
}

// ---------------------------------------------------------------------------
// WMMA GEMM: Out(nrows x NCOLS) = (A [+A2]) @ W + bias [+R], K = 128.
// One wave per 16x16 D tile; blockDim = 32*(NCOLS/16); grid = nrows/16.
// HAS_A2 / HAS_R / NCOLS are compile-time: no epilogue branches, row strides
// become instruction-immediate offsets, loads/stores collapse into clauses.
// ---------------------------------------------------------------------------
template <bool HAS_A2, bool HAS_R, int NCOLS>
__global__ void wmma_gemm_bias_kernel(const float* __restrict__ A,
                                      const float* __restrict__ A2,
                                      const _Float16* __restrict__ Wpk,
                                      const float* __restrict__ bias,
                                      const float* __restrict__ R,
                                      float* __restrict__ Out)
{
    __shared__ _Float16 As[16][136];                 // 272B rows: 16B-aligned
    const int tid = threadIdx.x;
    const int rowbase = blockIdx.x * 16;

    // Stage A tile (16 x 128): vectorized f32x4 -> f16x4 -> 8B LDS stores.
    for (int i = tid; i < 16 * 32; i += blockDim.x) {
        const int r = i >> 5, c4 = (i & 31) << 2;
        const float4 va = *(const float4*)(A + (size_t)(rowbase + r) * 128 + c4);
        float fx = va.x, fy = va.y, fz = va.z, fw = va.w;
        if (HAS_A2) {
            const float4 vb = *(const float4*)(A2 + (size_t)(rowbase + r) * 128 + c4);
            fx += vb.x; fy += vb.y; fz += vb.z; fw += vb.w;
        }
        v4h h;
        h[0] = (_Float16)fx; h[1] = (_Float16)fy;
        h[2] = (_Float16)fz; h[3] = (_Float16)fw;
        *(v4h*)&As[r][c4] = h;
    }
    __syncthreads();

    const int wave = tid >> 5;
    const int lane = tid & 31;
    const int n    = wave * 16 + (lane & 15);
    const int kofs = (lane >= 16) ? 8 : 0;
    const int m    = lane & 15;

    // B fragments: one 32B coalesced vector load per K chunk.
    const v16h* wfrag = (const v16h*)Wpk + (size_t)(wave * 4) * 32 + lane;
    v16h bfrag[4];
#pragma unroll
    for (int kc = 0; kc < 4; ++kc) bfrag[kc] = wfrag[(size_t)kc * 32];

    v8f acc = {0.f, 0.f, 0.f, 0.f, 0.f, 0.f, 0.f, 0.f};
#pragma unroll
    for (int kc = 0; kc < 4; ++kc) {
        // A fragment: two aligned 16B LDS loads (K runs {0..7} and {16..23}+kofs)
        const v8h lo = *(const v8h*)&As[m][kc * 32 + kofs];
        const v8h hi = *(const v8h*)&As[m][kc * 32 + 16 + kofs];
        const v16h afrag = __builtin_shufflevector(lo, hi,
            0, 1, 2, 3, 4, 5, 6, 7, 8, 9, 10, 11, 12, 13, 14, 15);
        acc = __builtin_amdgcn_wmma_f32_16x16x32_f16(
            false, afrag, false, bfrag[kc], (short)0, acc, false, false);
    }

    const float bcol = bias[n];
    // Single base; per-row offsets r*NCOLS are compile-time immediates.
    float* obase = Out + (size_t)(rowbase + kofs) * NCOLS + n;   // row = r+kofs
    const float* rbase = R + (size_t)(rowbase + kofs) * NCOLS + n;

    float res[8];
    if (HAS_R) {
#pragma unroll
        for (int r = 0; r < 8; ++r) res[r] = rbase[r * NCOLS];
    }
#pragma unroll
    for (int r = 0; r < 8; ++r) {
        float v = acc[r] + bcol;
        if (HAS_R) v += res[r];
        obase[r * NCOLS] = v;
    }
}

// ---------------------------------------------------------------------------
// Sampler: one wave per query. Lane=(head,point) computes softmax attention
// (camera-independent) and bilinear corners; corner (weight,index) pairs are
// broadcast with v_readlane into SGPRs so the zero-weight skip is pure scalar
// flow and the gathers use SGPR-base addressing. Lane = channel within head
// -> 128B coalesced gathers from L2-resident valproj (13 MB << 192 MB L2).
// ---------------------------------------------------------------------------
__device__ __forceinline__ int readlane_i32(int v, int lane) {
    return __builtin_amdgcn_readlane(v, lane);
}

__global__ void sampler_kernel(const float* __restrict__ valproj,   // S*M x C
                               const float* __restrict__ projq,     // N x 96
                               const float* __restrict__ refpts,    // S*N*4*2
                               const int*   __restrict__ bevmask,   // S*N*4
                               float* __restrict__ slots)           // N x C
{
    const int lane = threadIdx.x & 31;
    const int q = blockIdx.x * (blockDim.x >> 5) + (threadIdx.x >> 5);
    if (q >= N_) return;                      // wave-uniform

    const int h = lane >> 3;                  // head 0..3
    const int p = lane & 7;                   // point 0..7
    const int z = p & 3;                      // Z anchor (p = i*Z + z)

    // ---- camera-independent: softmax over NP=8 within each head ----
    const float* qrow = projq + (size_t)q * 96;
    float lg = qrow[64 + lane];               // logits: col 64 + (h*8+p)
    float mx = lg;
    mx = fmaxf(mx, __shfl_xor(mx, 1));
    mx = fmaxf(mx, __shfl_xor(mx, 2));
    mx = fmaxf(mx, __shfl_xor(mx, 4));
    float ex = __expf(lg - mx);
    float sm = ex;
    sm += __shfl_xor(sm, 1);
    sm += __shfl_xor(sm, 2);
    sm += __shfl_xor(sm, 4);
    const float attnw = ex / sm;

    // normalized sampling offsets (divide by (W,H) per reference)
    const float offx = qrow[h * 16 + p * 2 + 0] * (1.0f / W_);
    const float offy = qrow[h * 16 + p * 2 + 1] * (1.0f / H_);

    float acc[4] = {0.f, 0.f, 0.f, 0.f};
    int count = 0;

    for (int s = 0; s < S_; ++s) {
        const int* mrow = bevmask + ((size_t)s * N_ + q) * 4;
        if (s + 1 < S_)   // gfx1250 global_prefetch for next camera's mask row
            __builtin_prefetch(bevmask + ((size_t)(s + 1) * N_ + q) * 4, 0, 1);
        const int valid = mrow[0] | mrow[1] | mrow[2] | mrow[3];  // mask in {0,1}
        if (!valid) continue;                 // wave-uniform
        ++count;

        const float* rp = refpts + (((size_t)s * N_ + q) * 4 + z) * 2;
        const float x = (rp[0] + offx) * W_ - 0.5f;
        const float y = (rp[1] + offy) * H_ - 0.5f;
        const float x0 = floorf(x), y0 = floorf(y);
        const float wx = x - x0,   wy = y - y0;
        const int ix0 = (int)x0,   iy0 = (int)y0;

        int cwbits[4];
        int cidx[4];
#pragma unroll
        for (int k = 0; k < 4; ++k) {
            const int dx = k & 1, dy = k >> 1;
            const int xi = ix0 + dx, yi = iy0 + dy;
            const bool inb = (xi >= 0) & (xi < W_) & (yi >= 0) & (yi < H_);
            const int xc = min(max(xi, 0), W_ - 1);
            const int yc = min(max(yi, 0), H_ - 1);
            cidx[k] = yc * W_ + xc;
            const float wgt = (dx ? wx : 1.f - wx) * (dy ? wy : 1.f - wy);
            cwbits[k] = inb ? __float_as_int(wgt * attnw) : 0;
        }

        const float* vbase = valproj + (size_t)s * M_ * C_;
#pragma unroll
        for (int head = 0; head < 4; ++head) {
            for (int pp = 0; pp < 8; ++pp) {
                const int hp = head * 8 + pp;
#pragma unroll
                for (int k = 0; k < 4; ++k) {
                    const int wb = readlane_i32(cwbits[k], hp);  // SGPR
                    if (wb != 0) {                               // scalar branch
                        const int   ix = readlane_i32(cidx[k], hp);  // SGPR
                        const float w  = __int_as_float(wb);
                        acc[head] += w * vbase[(size_t)ix * C_ + head * HD_ + lane];
                    }
                }
            }
        }
    }

    const float inv = 1.0f / (float)max(count, 1);
#pragma unroll
    for (int head = 0; head < 4; ++head)
        slots[(size_t)q * C_ + head * HD_ + lane] = acc[head] * inv;
}

// ---------------------------------------------------------------------------
extern "C" void kernel_launch(void* const* d_in, const int* in_sizes, int n_in,
                              void* d_out, int out_size, void* d_ws, size_t ws_size,
                              hipStream_t stream)
{
    (void)in_sizes; (void)n_in; (void)out_size; (void)ws_size;

    const float* query  = (const float*)d_in[0];
    // d_in[1] = key : unused by the reference computation
    const float* value  = (const float*)d_in[2];   // (S,M,1,C) == (S,M,C)
    const float* qpos   = (const float*)d_in[3];
    const float* refpts = (const float*)d_in[4];   // (S,1,N,Z,2)
    // d_in[5] = spatial_shapes : constants [H,W] baked in
    const int*   bevmsk = (const int*)d_in[6];     // (S,1,N,Z)
    const float* Wv = (const float*)d_in[7];
    const float* bv = (const float*)d_in[8];
    const float* Wo = (const float*)d_in[9];
    const float* bo = (const float*)d_in[10];
    const float* Wa = (const float*)d_in[11];
    const float* ba = (const float*)d_in[12];
    const float* Wp = (const float*)d_in[13];
    const float* bp = (const float*)d_in[14];

    float* out = (float*)d_out;

    // Workspace layout: f32 buffers, then 32B-aligned f16 packed weights.
    float* ws      = (float*)d_ws;
    float* valproj = ws;                                   // S*M*C = 3,244,032
    float* projq   = valproj + (size_t)S_ * M_ * C_;       // N*96  =   960,000
    float* slots   = projq   + (size_t)N_ * 96;            // N*C   = 1,280,000
    float* biasq   = slots   + (size_t)N_ * C_;            // 96 (pad to 128)
    _Float16* hbase = (_Float16*)(biasq + 128);            // 32B-aligned offset
    _Float16* Wvpk = hbase;                                // 128*128 halves
    _Float16* Wqpk = Wvpk + 128 * 128;                     // 128*96 (Wo|Wa)
    _Float16* Wppk = Wqpk + 128 * 96;                      // 128*128

    // 0) pack weights to f16 fragment order (tiny one-time kernels)
    pack_weights_kernel<<<(128 * 128) / 256, 256, 0, stream>>>(Wv, Wvpk, 128);
    pack_weights_kernel<<<(128 *  64) / 256, 256, 0, stream>>>(Wo, Wqpk, 64);
    pack_weights_kernel<<<(128 *  32) / 256, 256, 0, stream>>>(Wa, Wqpk + 128 * 64, 32);
    pack_weights_kernel<<<(128 * 128) / 256, 256, 0, stream>>>(Wp, Wppk, 128);
    concat_bias_kernel<<<1, 96, 0, stream>>>(bo, ba, biasq);

    // 1) value projection: (25344 x 128) @ (128 x 128) + bv
    wmma_gemm_bias_kernel<false, false, 128><<<(S_ * M_) / 16, 256, 0, stream>>>(
        value, nullptr, Wvpk, bv, nullptr, valproj);

    // 2) fused offsets+logits: (10000 x 128) @ (128 x 96) + [bo|ba]
    wmma_gemm_bias_kernel<true, false, 96><<<N_ / 16, 192, 0, stream>>>(
        query, qpos, Wqpk, biasq, nullptr, projq);

    // 3) deformable sampling + masked camera mean (8 queries per 256-thr block)
    sampler_kernel<<<N_ / 8, 256, 0, stream>>>(
        valproj, projq, refpts, bevmsk, slots);

    // 4) output projection + bias + residual (query) -> d_out
    wmma_gemm_bias_kernel<false, true, 128><<<N_ / 16, 256, 0, stream>>>(
        slots, nullptr, Wppk, bp, query, out);
}
